// Self_Attn_52690658787635
// MI455X (gfx1250) — matrix operand order repaired and checked by reading, not compiled, not run on hardware
//
#include <hip/hip_runtime.h>
#include <math.h>

// ---------------------------------------------------------------------------
// Self-attention (SAGAN-style) for MI455X / gfx1250.
// B=16, C=256, H=W=48 -> N=2304, d_k = C/8 = 32.
// All heavy matmuls run through v_wmma_f32_16x16x32_f16 (wave32 WMMA).
// Round 2: 32 queries per wave (2 P tiles share every K/V operand load),
// halving L2 traffic for K/V; scale folded into qh.
// ---------------------------------------------------------------------------

#define B_  16
#define C_  256
#define N_  2304
#define CK_ 32

typedef __attribute__((ext_vector_type(16))) _Float16 v16h;
typedef __attribute__((ext_vector_type(8)))  _Float16 v8h;
typedef __attribute__((ext_vector_type(8)))  float    v8f;

// ---- WMMA operand loaders (wave32, 16x16x32 f16 shapes) -------------------
// A matrix 16x32 (MxK), row-major source, row stride rs (f16 elems).
// Lanes 0-15: M=lane, K={0..7,16..23}; lanes 16-31: M=lane-16, K={8..15,24..31}.
__device__ __forceinline__ v16h load_matA(const _Float16* base, int rs) {
    int lane = threadIdx.x & 31;
    int h = lane >> 4, m = lane & 15;
    const _Float16* p = base + m * rs + h * 8;
    v8h lo = *(const v8h*)(p);        // K = h*8 .. h*8+7
    v8h hi = *(const v8h*)(p + 16);   // K = 16+h*8 .. 16+h*8+7
    v16h r;
#pragma unroll
    for (int i = 0; i < 8; i++) { r[i] = lo[i]; r[i + 8] = hi[i]; }
    return r;
}

// B matrix 32x16 (KxN) loaded from an [N][K] (col-of-B major) array,
// row stride rs. Lane n holds column n; lanes 0-15 K={0..7,16..23}, lanes 16-31
// K={8..15,24..31} (same K order as load_matA).
__device__ __forceinline__ v16h load_matB(const _Float16* base, int rs) {
    int lane = threadIdx.x & 31;
    int h = lane >> 4, n = lane & 15;
    const _Float16* p = base + n * rs + h * 8;
    v8h lo = *(const v8h*)(p);
    v8h hi = *(const v8h*)(p + 16);
    v16h r;
#pragma unroll
    for (int i = 0; i < 8; i++) { r[i] = lo[i]; r[i + 8] = hi[i]; }
    return r;
}

__device__ __forceinline__ v8f wmma_f16(v16h a, v16h b, v8f c) {
    return __builtin_amdgcn_wmma_f32_16x16x32_f16(
        false, a, false, b, (short)0, c, false, false);
}

// ---------------------------------------------------------------------------
// Kernel 1: convert Wq/Wk/Wv (f32) -> f16 copies in workspace.
// ---------------------------------------------------------------------------
__global__ void k_convert_w(const float* __restrict__ Wq,
                            const float* __restrict__ Wk,
                            const float* __restrict__ Wv,
                            _Float16* whq, _Float16* whk, _Float16* whv) {
    int idx = blockIdx.x * blockDim.x + threadIdx.x;
    const int nq = CK_ * C_;   // 8192
    const int nv = C_ * C_;    // 65536
    if (idx < nq)                whq[idx]           = (_Float16)Wq[idx];
    else if (idx < 2 * nq)       whk[idx - nq]      = (_Float16)Wk[idx - nq];
    else if (idx < 2 * nq + nv)  whv[idx - 2 * nq]  = (_Float16)Wv[idx - 2 * nq];
}

// ---------------------------------------------------------------------------
// Kernel 2: xh[b][n][c] = f16(x[b][c][n])  (tiled LDS transpose, f32->f16)
// ---------------------------------------------------------------------------
__global__ void k_transpose_x(const float* __restrict__ x, _Float16* __restrict__ xh) {
    __shared__ float t[16][17];
    int n0 = blockIdx.x * 16, c0 = blockIdx.y * 16, b = blockIdx.z;
    int tx = threadIdx.x, ty = threadIdx.y;
    t[ty][tx] = x[((size_t)(b * C_ + c0 + ty)) * N_ + n0 + tx];
    __syncthreads();
    xh[((size_t)(b * N_ + n0 + ty)) * C_ + c0 + tx] = (_Float16)t[tx][ty];
}

// ---------------------------------------------------------------------------
// Kernel 3: fused q/k/v projection, one wave per 16x16 output tile.
// Row tiles 0..15 -> Wv (store vh[b][c][n]); 16..17 -> Wq (store qh[b][n][d],
// with 1/sqrt(C) pre-folded); 18..19 -> Wk (store kh[b][n][d]).
// ---------------------------------------------------------------------------
__global__ void __launch_bounds__(32) k_project(
    const _Float16* __restrict__ xh,
    const _Float16* __restrict__ whq, const _Float16* __restrict__ whk,
    const _Float16* __restrict__ whv,
    const float* __restrict__ bq, const float* __restrict__ bk,
    const float* __restrict__ bv,
    _Float16* __restrict__ qh, _Float16* __restrict__ kh,
    _Float16* __restrict__ vh) {
    int bid = blockIdx.x;
    int nt = bid % (N_ / 16); bid /= (N_ / 16);
    int rt = bid % 20;        int b = bid / 20;
    int n0 = nt * 16;
    int lane = threadIdx.x & 31;
    int h = lane >> 4, nl = lane & 15;

    const _Float16* wbase; const float* bias; int row0;
    if (rt < 16)      { wbase = whv; bias = bv; row0 = rt * 16; }
    else if (rt < 18) { wbase = whq; bias = bq; row0 = (rt - 16) * 16; }
    else              { wbase = whk; bias = bk; row0 = (rt - 18) * 16; }

    const _Float16* xb = xh + (size_t)b * N_ * C_ + (size_t)n0 * C_;
    v8f acc = {};
#pragma unroll
    for (int k0 = 0; k0 < C_; k0 += 32) {
        v16h a  = load_matA(wbase + row0 * C_ + k0, C_);
        v16h bm = load_matB(xb + k0, C_);
        acc = wmma_f16(a, bm, acc);
    }
    if (rt < 16) {
        // vh[b][c][n] : lane holds column n, rows c vary over accumulator regs
#pragma unroll
        for (int r = 0; r < 8; r++) {
            int c = row0 + h * 8 + r;
            float val = acc[r] + bias[c];
            vh[((size_t)b * C_ + c) * N_ + n0 + nl] = (_Float16)val;
        }
    } else {
        // qh/kh stored transposed as [b][n][32]: 8 contiguous f16 per lane.
        // q gets the 1/sqrt(C)=1/16 softmax scale folded in here.
        _Float16* dst = (rt < 18) ? qh : kh;
        float s = (rt < 18) ? 0.0625f : 1.0f;
        v8h pk;
#pragma unroll
        for (int r = 0; r < 8; r++) {
            int d = row0 + h * 8 + r;
            pk[r] = (_Float16)((acc[r] + bias[d]) * s);
        }
        *(v8h*)(dst + ((size_t)b * N_ + n0 + nl) * CK_ + row0 + h * 8) = pk;
    }
}

// ---------------------------------------------------------------------------
// Kernel 4: flash-attention. Block = 4 waves; block covers a 32-query tile
// across all 256 channels. Each wave: 32 queries x 64 channels.
// Per 32-key chunk: 1 K-operand pair shared by both query tiles (4 energy
// WMMAs), online softmax per tile, LDS C->A relayout of P, 4 shared V
// operands feeding 8 PV WMMAs.  72 chunks over N=2304.
// ---------------------------------------------------------------------------
__global__ void __launch_bounds__(128) k_attn(
    const _Float16* __restrict__ qh, const _Float16* __restrict__ kh,
    const _Float16* __restrict__ vh, const float* __restrict__ x,
    const float* __restrict__ gamma, float* __restrict__ out) {
    __shared__ _Float16 lp[8][16 * 32];   // per-(wave, query-tile) P staging

    int bid = blockIdx.x;
    int it = bid % (N_ / 32); int b = bid / (N_ / 32);
    int i0 = it * 32;
    int w = threadIdx.x >> 5;
    int lane = threadIdx.x & 31;
    int h = lane >> 4, nl = lane & 15;
    int c0 = w * 64;

    // Two Q tiles (16 x 32 each) -- A operands, reused for all key chunks
    v16h qa0 = load_matA(qh + ((size_t)b * N_ + i0)      * CK_, CK_);
    v16h qa1 = load_matA(qh + ((size_t)b * N_ + i0 + 16) * CK_, CK_);

    v8f acc[2][4] = {{{}, {}, {}, {}}, {{}, {}, {}, {}}};
    float mrun[2][8], lrun[2][8];
#pragma unroll
    for (int qt = 0; qt < 2; qt++)
#pragma unroll
        for (int r = 0; r < 8; r++) { mrun[qt][r] = -1e30f; lrun[qt][r] = 0.f; }

    const _Float16* kb = kh + (size_t)b * N_ * CK_;
    const _Float16* vb = vh + (size_t)b * C_ * N_;

    for (int j0 = 0; j0 < N_; j0 += 32) {
        v16h bk0 = load_matB(kb + (size_t)j0 * CK_, CK_);
        v16h bk1 = load_matB(kb + (size_t)(j0 + 16) * CK_, CK_);

#pragma unroll
        for (int qt = 0; qt < 2; qt++) {
            v16h qa = qt ? qa1 : qa0;
            v8f z = {};
            v8f s0 = wmma_f16(qa, bk0, z);   // scale pre-folded into q
            v8f s1 = wmma_f16(qa, bk1, z);

            float p0[8], p1[8], alpha[8];
#pragma unroll
            for (int r = 0; r < 8; r++) {
                float a = s0[r], bb = s1[r];
                float cm = fmaxf(a, bb);
                cm = fmaxf(cm, __shfl_xor(cm, 1));
                cm = fmaxf(cm, __shfl_xor(cm, 2));
                cm = fmaxf(cm, __shfl_xor(cm, 4));
                cm = fmaxf(cm, __shfl_xor(cm, 8));
                float mn = fmaxf(mrun[qt][r], cm);
                alpha[r] = __expf(mrun[qt][r] - mn);
                mrun[qt][r] = mn;
                p0[r] = __expf(a - mn);
                p1[r] = __expf(bb - mn);
                float rs = p0[r] + p1[r];
                rs += __shfl_xor(rs, 1);
                rs += __shfl_xor(rs, 2);
                rs += __shfl_xor(rs, 4);
                rs += __shfl_xor(rs, 8);
                lrun[qt][r] = lrun[qt][r] * alpha[r] + rs;
            }
#pragma unroll
            for (int t = 0; t < 4; t++)
#pragma unroll
                for (int r = 0; r < 8; r++) acc[qt][t][r] *= alpha[r];

            // stage P (16 queries x 32 keys, row-major f16) into LDS
            _Float16* lw = &lp[w * 2 + qt][0];
#pragma unroll
            for (int r = 0; r < 8; r++) {
                int m = h * 8 + r;
                lw[m * 32 + nl]      = (_Float16)p0[r];
                lw[m * 32 + 16 + nl] = (_Float16)p1[r];
            }
        }
        asm volatile("s_wait_dscnt 0x0" ::: "memory");

        v16h pa0 = load_matA(&lp[w * 2 + 0][0], 32);
        v16h pa1 = load_matA(&lp[w * 2 + 1][0], 32);
#pragma unroll
        for (int t = 0; t < 4; t++) {
            v16h bv = load_matB(vb + (size_t)(c0 + t * 16) * N_ + j0, N_);
            acc[0][t] = wmma_f16(pa0, bv, acc[0][t]);
            acc[1][t] = wmma_f16(pa1, bv, acc[1][t]);
        }
    }

    float g = gamma[0];
#pragma unroll
    for (int qt = 0; qt < 2; qt++)
#pragma unroll
        for (int t = 0; t < 4; t++) {
            int c = c0 + t * 16 + nl;
#pragma unroll
            for (int r = 0; r < 8; r++) {
                int i = i0 + qt * 16 + h * 8 + r;
                size_t idx = ((size_t)b * C_ + c) * N_ + i;
                out[idx] = g * (acc[qt][t][r] / lrun[qt][r]) + x[idx];
            }
        }
}

// ---------------------------------------------------------------------------
// Launcher
// ---------------------------------------------------------------------------
extern "C" void kernel_launch(void* const* d_in, const int* in_sizes, int n_in,
                              void* d_out, int out_size, void* d_ws, size_t ws_size,
                              hipStream_t stream) {
    const float* x     = (const float*)d_in[0];
    const float* Wq    = (const float*)d_in[1];
    const float* bq    = (const float*)d_in[2];
    const float* Wk    = (const float*)d_in[3];
    const float* bk    = (const float*)d_in[4];
    const float* Wv    = (const float*)d_in[5];
    const float* bv    = (const float*)d_in[6];
    const float* gamma = (const float*)d_in[7];
    float* out = (float*)d_out;

    // Workspace layout (bytes), ~40.7 MB total
    char* ws = (char*)d_ws;
    _Float16* xh  = (_Float16*)(ws);              // 16*2304*256*2 = 18,874,368
    _Float16* whq = (_Float16*)(ws + 18874368);   //   16,384
    _Float16* whk = (_Float16*)(ws + 18890752);   //   16,384
    _Float16* whv = (_Float16*)(ws + 18907136);   //  131,072
    _Float16* qh  = (_Float16*)(ws + 19038208);   // 2,359,296
    _Float16* kh  = (_Float16*)(ws + 21397504);   // 2,359,296
    _Float16* vh  = (_Float16*)(ws + 23756800);   // 18,874,368

    k_convert_w<<<(2 * CK_ * C_ + C_ * C_ + 255) / 256, 256, 0, stream>>>(
        Wq, Wk, Wv, whq, whk, whv);

    dim3 tg(N_ / 16, C_ / 16, B_);
    k_transpose_x<<<tg, dim3(16, 16), 0, stream>>>(x, xh);

    k_project<<<B_ * 20 * (N_ / 16), 32, 0, stream>>>(
        xh, whq, whk, whv, bq, bk, bv, qh, kh, vh);

    k_attn<<<B_ * (N_ / 32), 128, 0, stream>>>(qh, kh, vh, x, gamma, out);
}
